// TemporalVoxelAttention_58789512348258
// MI455X (gfx1250) — matrix operand order repaired and checked by reading, not compiled, not run on hardware
//
#include <hip/hip_runtime.h>
#include <hip/hip_bf16.h>

// ---------------- problem constants ----------------
#define NVOX 100000
#define PPTS 48
#define TV   16            // voxels per workgroup
#define NBLK (NVOX / TV)   // 6250, exact

typedef __attribute__((ext_vector_type(16))) _Float16 v16h;
typedef __attribute__((ext_vector_type(8)))  _Float16 v8h;
typedef __attribute__((ext_vector_type(8)))  float    v8f;
typedef __attribute__((ext_vector_type(4)))  unsigned int u32x4;
typedef __attribute__((ext_vector_type(8)))  int i32x8;
typedef __attribute__((ext_vector_type(4)))  int i32x4;

// f16 weight offsets inside d_ws / sW (in halves)
#define OFF_PPW1 0        // 64x8
#define OFF_PPW2 512      // 64x64
#define OFF_INPJ 4608     // 192x64
#define OFF_OUTW 16896    // 64x64
#define OFF_FFN1 20992    // 128x64
#define OFF_FFN2 29184    // 64x128
#define OFF_OPW1 37376    // 64x64
#define OFF_OPW2 41472    // 64x64
#define WTOT     45568    // total halves (91,136 B)

// LDS leading dims (elements)
#define TLD 72    // tokens buffers (48 rows x 64 cols)
#define QLD 200   // qkv f32 (48 x 192)
#define FLD 136   // ffn hidden f16 (48 x 128)

#if __has_builtin(__builtin_amdgcn_tensor_load_to_lds)
#define USE_TDM 1
#else
#define USE_TDM 0
#endif

#define WMMA(a, b, c) \
  __builtin_amdgcn_wmma_f32_16x16x32_f16(false, (a), false, (b), (short)0, (c), false, false)

// A fragment, 16x32 f16 row-major, contiguous K layout:
// lanes 0-15 hold K=0..15 contiguously, lanes 16-31 hold K=16..31
__device__ __forceinline__ v16h frag_ld_a(const _Float16* base, int ld, int lane) {
  const _Float16* p = base + (lane & 15) * ld + ((lane & 16) ? 16 : 0);
  v8h lo = *(const v8h*)(p);
  v8h hi = *(const v8h*)(p + 8);
  v16h f;
#pragma unroll
  for (int i = 0; i < 8; ++i) { f[i] = lo[i]; f[i + 8] = hi[i]; }
  return f;
}

// B fragment for X @ W^T : W is (out,in) row-major f16; column n = lane&15.
// lanes 0-15 hold K=0..15 contiguously, lanes 16-31 hold K=16..31.
__device__ __forceinline__ v16h frag_ld_b(const _Float16* base, int ld, int lane) {
  const _Float16* p = base + (lane & 15) * ld + ((lane & 16) ? 16 : 0);
  v8h lo = *(const v8h*)(p);
  v8h hi = *(const v8h*)(p + 8);
  v16h f;
#pragma unroll
  for (int i = 0; i < 8; ++i) { f[i] = lo[i]; f[i + 8] = hi[i]; }
  return f;
}

__device__ __forceinline__ v8f bias_c(const float* bias, int col0, int lane) {
  float bv = bias[col0 + (lane & 15)];
  v8f c;
#pragma unroll
  for (int i = 0; i < 8; ++i) c[i] = bv;
  return c;
}

// D tile (16x16 f32): VGPR i -> M = i + 8*(lane>=16), N = lane&15
__device__ __forceinline__ void tile_st_f32(float* base, int ld, v8f d, int lane) {
  int n = lane & 15, mb = (lane & 16) ? 8 : 0;
#pragma unroll
  for (int i = 0; i < 8; ++i) base[(mb + i) * ld + n] = d[i];
}
__device__ __forceinline__ void tile_st_f16(_Float16* base, int ld, v8f d, int lane) {
  int n = lane & 15, mb = (lane & 16) ? 8 : 0;
#pragma unroll
  for (int i = 0; i < 8; ++i) base[(mb + i) * ld + n] = (_Float16)d[i];
}
__device__ __forceinline__ void tile_st_f16_relu(_Float16* base, int ld, v8f d, int lane) {
  int n = lane & 15, mb = (lane & 16) ? 8 : 0;
#pragma unroll
  for (int i = 0; i < 8; ++i) base[(mb + i) * ld + n] = (_Float16)fmaxf(d[i], 0.f);
}
__device__ __forceinline__ void tile_add_res(const float* base, int ld, v8f* d, int lane) {
  int n = lane & 15, mb = (lane & 16) ? 8 : 0;
#pragma unroll
  for (int i = 0; i < 8; ++i) (*d)[i] += base[(mb + i) * ld + n];
}

__device__ __forceinline__ float wred(float x) {
#pragma unroll
  for (int o = 16; o > 0; o >>= 1) x += __shfl_xor(x, o, 32);
  return x;
}

__device__ __forceinline__ void ln_row(float* rf, _Float16* rh,
                                       const float* g, const float* b) {
  float mu = 0.f;
#pragma unroll
  for (int c = 0; c < 64; ++c) mu += rf[c];
  mu *= (1.f / 64.f);
  float var = 0.f;
#pragma unroll
  for (int c = 0; c < 64; ++c) { float d = rf[c] - mu; var += d * d; }
  var *= (1.f / 64.f);
  float rs = rsqrtf(var + 1e-5f);
#pragma unroll
  for (int c = 0; c < 64; ++c) {
    float y = (rf[c] - mu) * rs * g[c] + b[c];
    rf[c] = y;
    rh[c] = (_Float16)y;
  }
}

// ---------------- weight f32 -> f16 prep ----------------
__global__ void cvt_f16_kernel(const float* __restrict__ src, _Float16* __restrict__ dst, int n) {
  int i = blockIdx.x * 256 + threadIdx.x;
  if (i < n) dst[i] = (_Float16)src[i];
}

// ---------------- fused transformer kernel ----------------
union BigU {
  float    qkv[48 * QLD];     // 38400 B
  _Float16 h1[32 * TLD];      // point-proj hidden
  _Float16 ffn[48 * FLD];     // ffn hidden
  _Float16 oph[16 * TLD];     // op hidden
};

__global__ __launch_bounds__(128) void tva_main(
    const float* __restrict__ vf, const float* __restrict__ vox,
    const int* __restrict__ npts, const _Float16* __restrict__ W,
    const float* __restrict__ pp_b1, const float* __restrict__ pp_b2,
    const float* __restrict__ in_b,  const float* __restrict__ out_b,
    const float* __restrict__ ln1g,  const float* __restrict__ ln1b,
    const float* __restrict__ ffb1,  const float* __restrict__ ffb2,
    const float* __restrict__ ln2g,  const float* __restrict__ ln2b,
    const float* __restrict__ opb1,  const float* __restrict__ opb2,
    float* __restrict__ out) {
  __shared__ __align__(16) float    sTokF[48 * TLD];   // tokens f32 (residual)
  __shared__ __align__(16) _Float16 sTokH[48 * TLD];   // tokens / ctx f16 (GEMM A)
  __shared__ __align__(16) BigU     uB;
  __shared__ __align__(16) _Float16 sW[WTOT];          // all weights, f16
  __shared__ __align__(16) _Float16 sCH16[32 * 8];     // cur/his means, K=8
  __shared__ int sHasCur[TV];
  __shared__ int sHasHis[TV];

  const int tid  = threadIdx.x;
  const int lane = tid & 31;
  const int wave = tid >> 5;
  const int v0   = blockIdx.x * TV;

  // ---- Stage 0w: TDM-stage all f16 weights into LDS (overlaps stage 0) ----
#if USE_TDM
  if (wave == 0) {
    unsigned lds_w = (unsigned)(unsigned long long)(void*)sW;
    unsigned long long ga = (unsigned long long)(const void*)W;
    // D# group0: count=1 | lds_addr | global_addr[56:0] | type=2
    u32x4 g0;
    g0[0] = 1u;
    g0[1] = lds_w;
    g0[2] = (unsigned)ga;
    g0[3] = ((unsigned)(ga >> 32) & 0x01FFFFFFu) | 0x80000000u;
    // D# group1: data_size=1 (2B); tensor_dim0=WTOT; tensor_dim1=1;
    //            tile_dim0=WTOT; tile_dim1=1; stride0=WTOT
    i32x8 g1;
    g1[0] = 0x00010000;                       // data_size=1 @ [17:16]
    g1[1] = (int)((WTOT & 0xFFFF) << 16);     // tensor_dim0[15:0] @ [63:48]
    g1[2] = (int)(((WTOT >> 16) & 0xFFFF) | (1 << 16)); // dim0 hi | tensor_dim1 lo
    g1[3] = (int)((WTOT & 0xFFFF) << 16);     // tile_dim0 @ [127:112]
    g1[4] = 1;                                // tile_dim1 = 1
    g1[5] = WTOT;                             // tensor_dim0_stride lo
    g1[6] = 0;
    g1[7] = 0;
    i32x4 g2; g2[0] = 0; g2[1] = 0; g2[2] = 0; g2[3] = 0;
    i32x4 g3 = g2;
#if __clang_major__ >= 23
    i32x8 g4; 
#pragma unroll
    for (int i = 0; i < 8; ++i) g4[i] = 0;
    __builtin_amdgcn_tensor_load_to_lds(g0, g1, g2, g3, g4, 0);
#else
    __builtin_amdgcn_tensor_load_to_lds(g0, g1, g2, g3, 0);
#endif
  }
#else
  // fallback: cooperative copy (aligned 16B chunks)
  for (int c = tid; c < WTOT / 8; c += 128)
    ((v8h*)sW)[c] = ((const v8h*)W)[c];
#endif

  // ---- Stage 0a: async DMA voxel_features -> sTokF rows 0..15 (f32) ----
  {
    const float* base = vf + (size_t)v0 * 64;
#pragma unroll
    for (int k = 0; k < 2; ++k) {
      int g = tid + k * 128;                 // 0..255 chunks of 16B
      int r = g >> 4, c4 = (g & 15) * 4;
      unsigned lds_off = (unsigned)(unsigned long long)(void*)&sTokF[r * TLD + c4];
      unsigned long long gaddr = (unsigned long long)(const void*)(base + r * 64 + c4);
      asm volatile("global_load_async_to_lds_b128 %0, %1, off"
                   :: "v"(lds_off), "v"(gaddr) : "memory");
    }
  }

  // ---- Stage 0b: masked means (one wave per voxel, 4 voxels/wave) ----
  for (int j = 0; j < 4; ++j) {
    int vL = wave * 4 + j;
    int v  = v0 + vL;
    int np = npts[v];
    float cs[8], hs[8];
#pragma unroll
    for (int i = 0; i < 8; ++i) { cs[i] = 0.f; hs[i] = 0.f; }
    float cc = 0.f, hc = 0.f;
#pragma unroll
    for (int rep = 0; rep < 2; ++rep) {
      int p = lane + rep * 32;
      if (p < PPTS) {
        const float4* pf4 = (const float4*)(vox + ((size_t)v * PPTS + p) * 8);
        __builtin_prefetch(pf4 + 2 * PPTS, 0, 1);  // next voxel's points
        float4 A = pf4[0], B = pf4[1];
        float pv[8] = {A.x, A.y, A.z, A.w, B.x, B.y, B.z, B.w};
        float valid = (p < np) ? 1.f : 0.f;
        float cm = (pv[7] > 0.5f) ? valid : 0.f;
        float hm = valid - cm;
#pragma unroll
        for (int i = 0; i < 8; ++i) { cs[i] += cm * pv[i]; hs[i] += hm * pv[i]; }
        cc += cm; hc += hm;
      }
    }
#pragma unroll
    for (int i = 0; i < 8; ++i) { cs[i] = wred(cs[i]); hs[i] = wred(hs[i]); }
    cc = wred(cc); hc = wred(hc);
    if (lane == 0) {
      float ic = 1.f / fmaxf(cc, 1.f), ih = 1.f / fmaxf(hc, 1.f);
#pragma unroll
      for (int i = 0; i < 8; ++i) {
        sCH16[vL * 8 + i]        = (_Float16)(cs[i] * ic);
        sCH16[(16 + vL) * 8 + i] = (_Float16)(hs[i] * ih);
      }
      sHasCur[vL] = (cc > 0.5f);
      sHasHis[vL] = (hc > 0.5f);
    }
  }

  // ---- complete async + tensor DMA, then publish to all waves ----
  asm volatile("s_wait_asynccnt 0x0" ::: "memory");
#if USE_TDM
#if __has_builtin(__builtin_amdgcn_s_wait_tensorcnt)
  __builtin_amdgcn_s_wait_tensorcnt(0);
#else
  asm volatile("s_wait_tensorcnt 0x0" ::: "memory");
#endif
#endif
  __syncthreads();

  // token0 f16 mirror (rows 0..15) — overlaps stage 1a below
  for (int idx = tid; idx < TV * 64; idx += 128) {
    int r = idx >> 6, c = idx & 63;
    sTokH[r * TLD + c] = (_Float16)sTokF[r * TLD + c];
  }

  // ---- Stage 1a: H1 = relu([cur;his](32x8) @ pp_w1^T + b1) -> 32x64 f16 ----
  for (int t = wave; t < 8; t += 4) {
    int mt = t >> 2, nt = t & 3;
    v16h a, b;
#pragma unroll
    for (int i = 0; i < 16; ++i) { a[i] = (_Float16)0.f; b[i] = (_Float16)0.f; }
    if (lane < 16) {  // only K=0..7 nonzero; K-pad to 32 with zeros
      v8h av = *(const v8h*)&sCH16[(mt * 16 + lane) * 8];
      v8h bv = *(const v8h*)&sW[OFF_PPW1 + (nt * 16 + lane) * 8];
#pragma unroll
      for (int i = 0; i < 8; ++i) { a[i] = av[i]; b[i] = bv[i]; }
    }
    v8f acc = bias_c(pp_b1, nt * 16, lane);
    acc = WMMA(a, b, acc);
    tile_st_f16_relu(&uB.h1[(mt * 16) * TLD + nt * 16], TLD, acc, lane);
  }
  __syncthreads();

  // ---- Stage 1b: tokens[1],[2] = H1 @ pp_w2^T + b2 -> rows 16..47 ----
  for (int t = wave; t < 8; t += 4) {
    int mt = t >> 2, nt = t & 3;
    v8f acc = bias_c(pp_b2, nt * 16, lane);
#pragma unroll
    for (int kt = 0; kt < 2; ++kt) {
      v16h a = frag_ld_a(&uB.h1[(mt * 16) * TLD + kt * 32], TLD, lane);
      v16h b = frag_ld_b(&sW[OFF_PPW2 + (nt * 16) * 64 + kt * 32], 64, lane);
      acc = WMMA(a, b, acc);
    }
    int r0 = 16 + mt * 16;  // token1 rows 16..31, token2 rows 32..47
    tile_st_f32(&sTokF[r0 * TLD + nt * 16], TLD, acc, lane);
    tile_st_f16(&sTokH[r0 * TLD + nt * 16], TLD, acc, lane);
  }
  __syncthreads();

  // ---- Stage 2: qkv = tokens(48x64) @ in_proj^T(64->192) + b ----
  for (int t = wave; t < 36; t += 4) {
    int mt = t / 12, nt = t % 12;
    v8f acc = bias_c(in_b, nt * 16, lane);
#pragma unroll
    for (int kt = 0; kt < 2; ++kt) {
      v16h a = frag_ld_a(&sTokH[(mt * 16) * TLD + kt * 32], TLD, lane);
      v16h b = frag_ld_b(&sW[OFF_INPJ + (nt * 16) * 64 + kt * 32], 64, lane);
      acc = WMMA(a, b, acc);
    }
    tile_st_f32(&uB.qkv[(mt * 16) * QLD + nt * 16], QLD, acc, lane);
  }
  __syncthreads();

  // ---- Stage 3: attention (3 tokens, 4 heads, DH=16), ctx -> sTokH f16 ----
  if (tid < 64) {
    int vL = tid >> 2, h = tid & 3;
    float pad1 = sHasCur[vL] ? 0.f : -1e30f;
    float pad2 = sHasHis[vL] ? 0.f : -1e30f;
    const float rsdh = 0.25f;  // 1/sqrt(16)
#pragma unroll
    for (int tq = 0; tq < 3; ++tq) {
      int qr = tq * 16 + vL;
      float qv[16];
#pragma unroll
      for (int d = 0; d < 16; ++d) qv[d] = uB.qkv[qr * QLD + h * 16 + d];
      float s[3];
#pragma unroll
      for (int k3 = 0; k3 < 3; ++k3) {
        int kr = k3 * 16 + vL;
        float acc = 0.f;
#pragma unroll
        for (int d = 0; d < 16; ++d) acc += qv[d] * uB.qkv[kr * QLD + 64 + h * 16 + d];
        s[k3] = acc * rsdh;
      }
      s[1] += pad1; s[2] += pad2;
      float m = fmaxf(s[0], fmaxf(s[1], s[2]));
      float e0 = __expf(s[0] - m), e1 = __expf(s[1] - m), e2 = __expf(s[2] - m);
      float inv = 1.f / (e0 + e1 + e2);
      e0 *= inv; e1 *= inv; e2 *= inv;
#pragma unroll
      for (int d = 0; d < 16; ++d) {
        int c = 128 + h * 16 + d;
        float cx = e0 * uB.qkv[vL * QLD + c] + e1 * uB.qkv[(16 + vL) * QLD + c] +
                   e2 * uB.qkv[(32 + vL) * QLD + c];
        sTokH[qr * TLD + h * 16 + d] = (_Float16)cx;  // ctx overwrites tokens-f16
      }
    }
  }
  __syncthreads();

  // ---- Stage 4: attn_out = ctx @ out_w^T + b ; += residual -> sTokF ----
  for (int t = wave; t < 12; t += 4) {
    int mt = t >> 2, nt = t & 3;
    v8f acc = bias_c(out_b, nt * 16, lane);
#pragma unroll
    for (int kt = 0; kt < 2; ++kt) {
      v16h a = frag_ld_a(&sTokH[(mt * 16) * TLD + kt * 32], TLD, lane);
      v16h b = frag_ld_b(&sW[OFF_OUTW + (nt * 16) * 64 + kt * 32], 64, lane);
      acc = WMMA(a, b, acc);
    }
    tile_add_res(&sTokF[(mt * 16) * TLD + nt * 16], TLD, &acc, lane);
    tile_st_f32(&sTokF[(mt * 16) * TLD + nt * 16], TLD, acc, lane);
  }
  __syncthreads();

  // ---- LN1 ----
  if (tid < 48) ln_row(&sTokF[tid * TLD], &sTokH[tid * TLD], ln1g, ln1b);
  __syncthreads();

  // ---- Stage 5a: FFN1 = relu(tokens @ ffn_w1^T + b1) -> 48x128 f16 ----
  for (int t = wave; t < 24; t += 4) {
    int mt = t >> 3, nt = t & 7;
    v8f acc = bias_c(ffb1, nt * 16, lane);
#pragma unroll
    for (int kt = 0; kt < 2; ++kt) {
      v16h a = frag_ld_a(&sTokH[(mt * 16) * TLD + kt * 32], TLD, lane);
      v16h b = frag_ld_b(&sW[OFF_FFN1 + (nt * 16) * 64 + kt * 32], 64, lane);
      acc = WMMA(a, b, acc);
    }
    tile_st_f16_relu(&uB.ffn[(mt * 16) * FLD + nt * 16], FLD, acc, lane);
  }
  __syncthreads();

  // ---- Stage 5b: FFN2 + residual -> sTokF ----
  for (int t = wave; t < 12; t += 4) {
    int mt = t >> 2, nt = t & 3;
    v8f acc = bias_c(ffb2, nt * 16, lane);
#pragma unroll
    for (int kt = 0; kt < 4; ++kt) {
      v16h a = frag_ld_a(&uB.ffn[(mt * 16) * FLD + kt * 32], FLD, lane);
      v16h b = frag_ld_b(&sW[OFF_FFN2 + (nt * 16) * 128 + kt * 32], 128, lane);
      acc = WMMA(a, b, acc);
    }
    tile_add_res(&sTokF[(mt * 16) * TLD + nt * 16], TLD, &acc, lane);
    tile_st_f32(&sTokF[(mt * 16) * TLD + nt * 16], TLD, acc, lane);
  }
  __syncthreads();

  // ---- LN2 ----
  if (tid < 48) ln_row(&sTokF[tid * TLD], &sTokH[tid * TLD], ln2g, ln2b);
  __syncthreads();

  // ---- Stage 6a: op hidden = relu(tok0 @ op_w1^T + b1) -> 16x64 f16 ----
  {
    int nt = wave;
    v8f acc = bias_c(opb1, nt * 16, lane);
#pragma unroll
    for (int kt = 0; kt < 2; ++kt) {
      v16h a = frag_ld_a(&sTokH[kt * 32], TLD, lane);  // rows 0..15 = token0
      v16h b = frag_ld_b(&sW[OFF_OPW1 + (nt * 16) * 64 + kt * 32], 64, lane);
      acc = WMMA(a, b, acc);
    }
    tile_st_f16_relu(&uB.oph[nt * 16], TLD, acc, lane);
  }
  __syncthreads();

  // ---- Stage 6b: refined = oph @ op_w2^T + b2 ; gate ; out = vf + refined ----
  {
    int nt = wave;
    v8f acc = bias_c(opb2, nt * 16, lane);
#pragma unroll
    for (int kt = 0; kt < 2; ++kt) {
      v16h a = frag_ld_a(&uB.oph[kt * 32], TLD, lane);
      v16h b = frag_ld_b(&sW[OFF_OPW2 + (nt * 16) * 64 + kt * 32], 64, lane);
      acc = WMMA(a, b, acc);
    }
    int n = lane & 15, mb = (lane & 16) ? 8 : 0;
#pragma unroll
    for (int i = 0; i < 8; ++i) {
      int vL = mb + i;
      size_t o = (size_t)(v0 + vL) * 64 + nt * 16 + n;
      float gate = sHasHis[vL] ? 1.f : 0.f;
      out[o] = vf[o] + acc[i] * gate;
    }
  }
}

extern "C" void kernel_launch(void* const* d_in, const int* in_sizes, int n_in,
                              void* d_out, int out_size, void* d_ws, size_t ws_size,
                              hipStream_t stream) {
  _Float16* wsh = (_Float16*)d_ws;
  auto cv = [&](int idx, int off, int n) {
    cvt_f16_kernel<<<(n + 255) / 256, 256, 0, stream>>>((const float*)d_in[idx], wsh + off, n);
  };
  cv(2,  OFF_PPW1, 64 * 8);     // pp_w1
  cv(4,  OFF_PPW2, 64 * 64);    // pp_w2
  cv(6,  OFF_INPJ, 192 * 64);   // in_proj_w
  cv(8,  OFF_OUTW, 64 * 64);    // out_w
  cv(12, OFF_FFN1, 128 * 64);   // ffn_w1
  cv(14, OFF_FFN2, 64 * 128);   // ffn_w2
  cv(18, OFF_OPW1, 64 * 64);    // op_w1
  cv(20, OFF_OPW2, 64 * 64);    // op_w2

  tva_main<<<NBLK, 128, 0, stream>>>(
      (const float*)d_in[0],   // voxel_features
      (const float*)d_in[1],   // voxels
      (const int*)d_in[22],    // num_points
      wsh,
      (const float*)d_in[3],   // pp_b1
      (const float*)d_in[5],   // pp_b2
      (const float*)d_in[7],   // in_proj_b
      (const float*)d_in[9],   // out_b
      (const float*)d_in[10],  // ln1_g
      (const float*)d_in[11],  // ln1_b
      (const float*)d_in[13],  // ffn_b1
      (const float*)d_in[15],  // ffn_b2
      (const float*)d_in[16],  // ln2_g
      (const float*)d_in[17],  // ln2_b
      (const float*)d_in[19],  // op_b1
      (const float*)d_in[21],  // op_b2
      (float*)d_out);
}
